// AttentionLayer_5952824672530
// MI455X (gfx1250) — matrix operand order repaired and checked by reading, not compiled, not run on hardware
//
#include <hip/hip_runtime.h>
#include <hip/hip_bf16.h>

// Problem dims (fixed by the reference)
#define BDIM 8
#define TDIM 1024
#define DDIM 1024
#define HDIM 16
#define ADIM 64
#define EDIM 262144

typedef __attribute__((ext_vector_type(16))) __bf16 v16bf;
typedef __attribute__((ext_vector_type(8)))  float  v8f;

union Frag16 { v16bf v; unsigned int u[8]; };

// f32 <-> bf16 via native hardware conversion
__device__ __forceinline__ unsigned short f2bf(float f) {
  union { __bf16 b; unsigned short u; } x; x.b = (__bf16)f; return x.u;
}
__device__ __forceinline__ float bf2f(unsigned short s) {
  union { __bf16 b; unsigned short u; } x; x.u = s; return (float)x.b;
}

// A-matrix (16x32, 16-bit) K base for VGPR j, lane-half hi.
// lanes 0-15: K = {0..7, 16..23}; lanes 16-31: K = {8..15, 24..31}; pairs (k,k+1) per VGPR.
__device__ __forceinline__ int kbaseA(int j, int hi) {
  return (j < 4) ? (hi * 8 + 2 * j) : (16 + hi * 8 + 2 * (j - 4));
}
// B-matrix (32x16, 16-bit): same K order as A: lanes 0-15 K = {0..7, 16..23},
// lanes 16-31 K = {8..15, 24..31}; pairs (k,k+1) per VGPR.
__device__ __forceinline__ int kbaseB(int j, int hi) {
  return (j < 4) ? (hi * 8 + 2 * j) : (16 + hi * 8 + 2 * (j - 4));
}

#if __has_builtin(__builtin_amdgcn_global_load_async_to_lds_b128)
#define HAVE_ASYNC_LDS 1
#endif

#define AS1 __attribute__((address_space(1)))
#define AS3 __attribute__((address_space(3)))
// gcc-style int4 vector, matching the builtin's parameter type exactly
typedef int gv4i __attribute__((vector_size(16)));

__device__ __forceinline__ void wait_asynccnt0() {
#if __has_builtin(__builtin_amdgcn_s_wait_asynccnt)
  __builtin_amdgcn_s_wait_asynccnt(0);
#else
  asm volatile("s_wait_asynccnt 0x0" ::: "memory");
#endif
}

// ---------------------------------------------------------------------------
// Generic C[M,N] = A[M,K] * B[K,N]  (f32 in, bf16 WMMA compute, f32/bf16 out)
// block = 128 threads (4 wave32), block tile 64(M) x 64(N), K step 32.
// ---------------------------------------------------------------------------
template <bool OUT_BF16>
__global__ __launch_bounds__(128)
void gemm_wmma(const float* __restrict__ A, const float* __restrict__ B,
               void* __restrict__ Cout, int M, int N, int K) {
  __shared__ unsigned short As[64 * 32];  // [m][k]
  __shared__ unsigned short Bs[64 * 32];  // [n][k] (transposed panel)

  const int tid   = threadIdx.x;
  const int lane  = tid & 31;
  const int wave  = tid >> 5;
  const int mlane = lane & 15;
  const int hi    = lane >> 4;
  const int mBlock = blockIdx.x * 64;
  const int nBlock = blockIdx.y * 64;

  v8f acc[4];
  for (int t = 0; t < 4; ++t)
    for (int i = 0; i < 8; ++i) acc[t][i] = 0.0f;

  for (int kc = 0; kc < K; kc += 32) {
    // prefetch next panels into cache while we stage/compute this one
    if (kc + 32 < K) {
      __builtin_prefetch(A + (size_t)(mBlock + (tid >> 1)) * K + kc + 32 + (tid & 1) * 16, 0, 1);
      __builtin_prefetch(B + (size_t)(kc + 32 + (tid >> 2)) * N + nBlock + (tid & 3) * 16, 0, 1);
    }
    // stage A panel 64x32 (row-major [m][k]) via float4 loads
    for (int i = 0; i < 4; ++i) {
      int e = i * 128 + tid;            // [0,512)
      int m = e >> 3, k4 = (e & 7) * 4;
      float4 v = *(const float4*)&A[(size_t)(mBlock + m) * K + kc + k4];
      unsigned short* d = &As[m * 32 + k4];
      d[0] = f2bf(v.x); d[1] = f2bf(v.y); d[2] = f2bf(v.z); d[3] = f2bf(v.w);
    }
    // stage B panel 32x64 transposed into [n][k]
    for (int i = 0; i < 4; ++i) {
      int e = i * 128 + tid;            // [0,512)
      int kk = e >> 4, n4 = (e & 15) * 4;
      float4 v = *(const float4*)&B[(size_t)(kc + kk) * N + nBlock + n4];
      Bs[(n4 + 0) * 32 + kk] = f2bf(v.x);
      Bs[(n4 + 1) * 32 + kk] = f2bf(v.y);
      Bs[(n4 + 2) * 32 + kk] = f2bf(v.z);
      Bs[(n4 + 3) * 32 + kk] = f2bf(v.w);
    }
    __syncthreads();

    Frag16 af;
    for (int j = 0; j < 8; ++j)
      af.u[j] = *(const unsigned int*)&As[(wave * 16 + mlane) * 32 + kbaseA(j, hi)];

    for (int t = 0; t < 4; ++t) {
      Frag16 bf;
      for (int j = 0; j < 8; ++j)
        bf.u[j] = *(const unsigned int*)&Bs[(t * 16 + mlane) * 32 + kbaseB(j, hi)];
      acc[t] = __builtin_amdgcn_wmma_f32_16x16x32_bf16(
          false, af.v, false, bf.v, (short)0, acc[t], false, false);
    }
    __syncthreads();
  }

  for (int t = 0; t < 4; ++t)
    for (int r = 0; r < 8; ++r) {
      int row = mBlock + wave * 16 + (hi ? r + 8 : r);
      int col = nBlock + t * 16 + mlane;
      if constexpr (OUT_BF16)
        ((unsigned short*)Cout)[(size_t)row * N + col] = f2bf(acc[t][r]);
      else
        ((float*)Cout)[(size_t)row * N + col] = acc[t][r];
    }
}

// ---------------------------------------------------------------------------
// Fused flash-attention over bf16 q/k/v: S = Q K^T + bs*ksum, scale, mask,
// online softmax, P V. block = 128 threads (4 waves) = one (b,h) x 64 q rows.
// K/V chunks staged to LDS with gfx1250 async-to-LDS when available.
// ---------------------------------------------------------------------------
__global__ __launch_bounds__(128)
void flash_attn_wmma(const unsigned short* __restrict__ qws,
                     const unsigned short* __restrict__ kws,
                     const unsigned short* __restrict__ vws,
                     const float* __restrict__ ksum,
                     const float* __restrict__ bsb,
                     const float* __restrict__ masks,
                     float* __restrict__ ctx) {
  __shared__ unsigned short Ks[32 * 64];     // [kk][a] natural
  __shared__ unsigned short Vn[32 * 64];     // [kk][a] natural
  __shared__ unsigned short Ps[4][16 * 32];  // per-wave P tile [m][kk]

  const int tid   = threadIdx.x;
  const int lane  = tid & 31;
  const int wave  = tid >> 5;
  const int mlane = lane & 15;
  const int hi    = lane >> 4;
  const int bh = blockIdx.x;
  const int b  = bh / HDIM;
  const int h  = bh % HDIM;
  const int qBlock = blockIdx.y * 64;
  const float scaleA = 0.125f;  // 1/sqrt(64)
  const float BIG = 1e30f;

  // Preload this wave's Q A-fragments straight from bf16 global (pairs contiguous in a)
  Frag16 qf[2];
  {
    const int qRow = qBlock + wave * 16 + mlane;  // A-frag M = lane&15
    const unsigned short* qb = qws + (((size_t)b * TDIM + qRow) * HDIM + h) * ADIM;
    for (int c = 0; c < 2; ++c)
      for (int j = 0; j < 8; ++j)
        qf[c].u[j] = *(const unsigned int*)&qb[c * 32 + kbaseA(j, hi)];
  }

  v8f cacc[4];
  for (int t = 0; t < 4; ++t)
    for (int i = 0; i < 8; ++i) cacc[t][i] = 0.0f;
  float mrow[8], lrow[8];
  for (int r = 0; r < 8; ++r) { mrow[r] = -3.0e38f; lrow[r] = 0.0f; }

  for (int kc = 0; kc < TDIM; kc += 32) {
    // ---- stage K and V chunks [32 kk][64 a] (raw bf16 copy) ----
#ifdef HAVE_ASYNC_LDS
    for (int i = 0; i < 2; ++i) {
      int e = i * 128 + tid;           // [0,256): 16B segments of the 4KB chunk
      int r = e >> 3, seg = (e & 7) * 8;
      size_t g = (((size_t)b * TDIM + kc + r) * HDIM + h) * ADIM + seg;
      __builtin_amdgcn_global_load_async_to_lds_b128(
          (AS1 gv4i*)(uintptr_t)(kws + g),
          (AS3 gv4i*)(unsigned)(uintptr_t)&Ks[r * 64 + seg], 0, 0);
      __builtin_amdgcn_global_load_async_to_lds_b128(
          (AS1 gv4i*)(uintptr_t)(vws + g),
          (AS3 gv4i*)(unsigned)(uintptr_t)&Vn[r * 64 + seg], 0, 0);
    }
    wait_asynccnt0();
#else
    for (int i = 0; i < 4; ++i) {
      int e = i * 128 + tid;           // [0,512): 8B segments
      int r = e >> 4, seg = (e & 15) * 4;
      size_t g = (((size_t)b * TDIM + kc + r) * HDIM + h) * ADIM + seg;
      *(uint2*)&Ks[r * 64 + seg] = *(const uint2*)&kws[g];
      *(uint2*)&Vn[r * 64 + seg] = *(const uint2*)&vws[g];
    }
#endif
    __syncthreads();

    // ---- S tiles: two 16-wide key tiles, contraction over a (2 x 32) ----
    v8f s[2];
    for (int t = 0; t < 2; ++t)
      for (int i = 0; i < 8; ++i) s[t][i] = 0.0f;
    for (int t = 0; t < 2; ++t)
      for (int c = 0; c < 2; ++c) {
        Frag16 bf;  // B[a][kk] = K[kk][a]; pair contiguous along a in Ks
        for (int j = 0; j < 8; ++j)
          bf.u[j] = *(const unsigned int*)&Ks[(t * 16 + mlane) * 64 + c * 32 + kbaseB(j, hi)];
        s[t] = __builtin_amdgcn_wmma_f32_16x16x32_bf16(
            false, qf[c].v, false, bf.v, (short)0, s[t], false, false);
      }

    // ---- bias * ksum, scale, mask ----
    float ksv[2];
    for (int t = 0; t < 2; ++t)
      ksv[t] = ksum[((size_t)b * TDIM + kc + t * 16 + mlane) * HDIM + h];
    for (int t = 0; t < 2; ++t)
      for (int r = 0; r < 8; ++r) {
        int trow = qBlock + wave * 16 + (hi ? r + 8 : r);
        size_t g = ((size_t)b * TDIM + trow) * TDIM + kc + t * 16 + mlane;
        s[t][r] = (s[t][r] + bsb[g] * ksv[t]) * scaleA - masks[g] * BIG;
      }

    // ---- online softmax (row reduce within 16-lane halves = rows r / r+8) ----
    for (int r = 0; r < 8; ++r) {
      float mx = fmaxf(s[0][r], s[1][r]);
      for (int d = 1; d < 16; d <<= 1) mx = fmaxf(mx, __shfl_xor(mx, d, 32));
      float mnew = fmaxf(mrow[r], mx);
      float corr = __expf(mrow[r] - mnew);
      mrow[r] = mnew;
      float p0 = __expf(s[0][r] - mnew);
      float p1 = __expf(s[1][r] - mnew);
      s[0][r] = p0; s[1][r] = p1;
      float rs = p0 + p1;
      for (int d = 1; d < 16; d <<= 1) rs += __shfl_xor(rs, d, 32);
      lrow[r] = lrow[r] * corr + rs;
      for (int t = 0; t < 4; ++t) cacc[t][r] *= corr;
    }

    // ---- P through LDS: C-layout -> A-layout (per-wave buffer) ----
    for (int t = 0; t < 2; ++t)
      for (int r = 0; r < 8; ++r) {
        int row = hi ? r + 8 : r;
        Ps[wave][row * 32 + t * 16 + mlane] = f2bf(s[t][r]);
      }
    __syncthreads();

    Frag16 pf;
    for (int j = 0; j < 8; ++j)
      pf.u[j] = *(const unsigned int*)&Ps[wave][mlane * 32 + kbaseA(j, hi)];

    // ---- P V: B[kk][a] from natural-layout Vn (pair = two rows) ----
    for (int t = 0; t < 4; ++t) {
      Frag16 vf;
      int col = t * 16 + mlane;
      for (int j = 0; j < 8; ++j) {
        int k0 = kbaseB(j, hi);
        unsigned int lo = Vn[k0 * 64 + col];
        unsigned int hp = Vn[(k0 + 1) * 64 + col];
        vf.u[j] = lo | (hp << 16);
      }
      cacc[t] = __builtin_amdgcn_wmma_f32_16x16x32_bf16(
          false, pf.v, false, vf.v, (short)0, cacc[t], false, false);
    }
    __syncthreads();
  }

  // normalize and store context [B,T,H,A] (f32 for the final projection GEMM)
  for (int r = 0; r < 8; ++r) {
    int trow = qBlock + wave * 16 + (hi ? r + 8 : r);
    float inv = 1.0f / lrow[r];
    for (int t = 0; t < 4; ++t) {
      int a = t * 16 + mlane;
      ctx[(((size_t)b * TDIM + trow) * HDIM + h) * ADIM + a] = cacc[t][r] * inv;
    }
  }
}

// ---------------------------------------------------------------------------
// helpers
// ---------------------------------------------------------------------------
__global__ void zero_f32_k(float* __restrict__ p, int n4) {
  int i = blockIdx.x * blockDim.x + threadIdx.x;
  if (i < n4) ((float4*)p)[i] = make_float4(0.f, 0.f, 0.f, 0.f);
}

__global__ void ksum_k(const unsigned short* __restrict__ k, float* __restrict__ ks) {
  int i = blockIdx.x * blockDim.x + threadIdx.x;  // [0, B*T*H)
  const unsigned short* p = k + (size_t)i * ADIM;
  float s = 0.f;
  for (int j = 0; j < ADIM; ++j) s += bf2f(p[j]);
  ks[i] = s;
}

__global__ void bias_scatter_k(const int* __restrict__ ab, const float* __restrict__ embs,
                               const float* __restrict__ scal, float* __restrict__ bs) {
  int e = blockIdx.x * blockDim.x + threadIdx.x;
  if (e >= EDIM) return;
  int et = ab[4 * e + 0];
  int bi = ab[4 * e + 1];
  int qi = ab[4 * e + 2];
  int ki = ab[4 * e + 3];
  float s = 0.f;
  for (int a = 0; a < ADIM; ++a) s += embs[et * ADIM + a] * scal[a];
  bs[((size_t)bi * TDIM + qi) * TDIM + ki] = s;
}

// ---------------------------------------------------------------------------
extern "C" void kernel_launch(void* const* d_in, const int* in_sizes, int n_in,
                              void* d_out, int out_size, void* d_ws, size_t ws_size,
                              hipStream_t stream) {
  (void)in_sizes; (void)n_in; (void)out_size; (void)ws_size;
  const float* states     = (const float*)d_in[0];
  const float* key_states = (const float*)d_in[1];
  const float* masks      = (const float*)d_in[2];
  const int*   ab         = (const int*)d_in[3];
  const float* Wq         = (const float*)d_in[4];
  const float* Wk         = (const float*)d_in[5];
  const float* Wv         = (const float*)d_in[6];
  const float* Wout       = (const float*)d_in[7];
  const float* embs       = (const float*)d_in[8];
  const float* scal       = (const float*)d_in[9];
  float* out = (float*)d_out;

  const size_t NQ  = (size_t)BDIM * TDIM * HDIM * ADIM;  // 8,388,608
  const size_t NBS = (size_t)BDIM * TDIM * TDIM;         // 8,388,608

  unsigned short* qws = (unsigned short*)d_ws;            // 16 MB
  unsigned short* kws = qws + NQ;                         // 16 MB
  unsigned short* vws = kws + NQ;                         // 16 MB
  float* ctx  = (float*)(vws + NQ);                       // 32 MB
  float* bsb  = ctx + NQ;                                 // 32 MB
  float* ksmb = bsb + NBS;                                // 0.5 MB

  const int M = BDIM * TDIM;   // 8192
  const int N = HDIM * ADIM;   // 1024
  const int K = DDIM;          // 1024

  // 1. dense bias buffer: zero then scatter
  zero_f32_k<<<(int)((NBS / 4 + 255) / 256), 256, 0, stream>>>(bsb, (int)(NBS / 4));
  bias_scatter_k<<<(EDIM + 255) / 256, 256, 0, stream>>>(ab, embs, scal, bsb);

  // 2. QKV projections (bf16 WMMA GEMMs, bf16 outputs)
  dim3 gg(M / 64, N / 64);
  gemm_wmma<true><<<gg, 128, 0, stream>>>(states,     Wq, (void*)qws, M, N, K);
  gemm_wmma<true><<<gg, 128, 0, stream>>>(key_states, Wk, (void*)kws, M, N, K);
  gemm_wmma<true><<<gg, 128, 0, stream>>>(key_states, Wv, (void*)vws, M, N, K);

  // 3. ksum[b,t,h]
  ksum_k<<<(BDIM * TDIM * HDIM) / 256, 256, 0, stream>>>(kws, ksmb);

  // 4. fused attention (async-to-LDS staging + WMMA)
  flash_attn_wmma<<<dim3(BDIM * HDIM, TDIM / 64), 128, 0, stream>>>(
      qws, kws, vws, ksmb, bsb, masks, ctx);

  // 5. output projection: [M, H*A] x [H*A, D] -> f32 out
  gemm_wmma<false><<<gg, 128, 0, stream>>>(ctx, Wout, (void*)out, M, DDIM, N);
}
